// FeatureFusionKAN_14912126452542
// MI455X (gfx1250) — matrix operand order repaired and checked by reading, not compiled, not run on hardware
//
#include <hip/hip_runtime.h>
#include <hip/hip_bf16.h>
#include <math.h>

// ---------------------------------------------------------------------------
// FeatureFusionKAN on MI455X (gfx1250): everything is GEMM -> bf16 WMMA with
// f32 accumulation, double-buffered LDS software pipeline (1 barrier / tile).
//   K1: attention GEMM (concat A) + sigmoid-gate epilogue -> kan_in f32, silu bf16
//   K2: base0 GEMM  (A = silu bf16)                        -> tmp f32
//   K3: spline0 GEMM (A = Cox-de Boor bases of kan_in)     -> h = tmp+acc, silu(h)
//   K4: base1 GEMM  (A = silu(h) bf16)                     -> tmp f32
//   K5: spline1 GEMM (A = bases of h)                      -> d_out = tmp+acc
// ---------------------------------------------------------------------------

typedef __attribute__((ext_vector_type(16))) __bf16 v16bf;
typedef __attribute__((ext_vector_type(8)))  __bf16 bf16x8;
typedef __attribute__((ext_vector_type(4)))  __bf16 bf16x4;
typedef __attribute__((ext_vector_type(8)))  float  v8f;

#define BM   128
#define BN   128
#define BK   32
#define LDT  40          // padded LDS row stride (bf16 elems): 80B -> conflict-free
#define FDIM 1024
#define BATCH 8192

// A-operand producers
#define A_CONCAT 0
#define A_BF16   1
#define A_SPLINE 2
// Epilogues
#define E_ATTN    0
#define E_PLAIN   1
#define E_SPLINE0 2
#define E_SPLINE1 3

__device__ __forceinline__ float sigmoidf_(float x) {
  // native v_rcp_f32 instead of IEEE divide (v_div_scale/fixup chain)
  return __builtin_amdgcn_rcpf(1.0f + __expf(-x));
}
__device__ __forceinline__ float siluf_(float x) { return x * sigmoidf_(x); }

// Cubic B-spline bases, efficient-kan grid: G=5, k=3, range [-1,1], h=0.4,
// knots t_j = -1 + (j-3)*0.4, j=0..11 -> 8 bases per x (Cox-de Boor,
// half-open intervals; out-of-range -> 0). Matches the reference recursion.
__device__ __forceinline__ void bspline8(float x, float* __restrict__ out) {
  const float h = 0.4f;
  float b[11];
#pragma unroll
  for (int j = 0; j < 11; ++j) {
    float t0 = -1.0f + (float)(j - 3) * h;
    b[j] = (x >= t0 && x < t0 + h) ? 1.0f : 0.0f;
  }
#pragma unroll
  for (int k = 1; k <= 3; ++k) {
    float inv = 1.0f / ((float)k * h);
#pragma unroll
    for (int j = 0; j < 10; ++j) {
      if (j < 11 - k) {
        float tj    = -1.0f + (float)(j - 3) * h;
        float left  = (x - tj) * inv;
        float right = ((tj + (float)(k + 1) * h) - x) * inv;
        b[j] = left * b[j] + right * b[j + 1];
      }
    }
  }
#pragma unroll
  for (int j = 0; j < 8; ++j) out[j] = b[j];
}

// A fragment (16x32 bf16): lane<16 -> M=lane, K = {0..7, 16..23};
// lane>=16 -> M=lane-16, K = {8..15, 24..31}. p already offset by half*8.
__device__ __forceinline__ v16bf frag_a(const __bf16* p) {
  union { v16bf v; bf16x8 h[2]; } u;
  u.h[0] = *(const bf16x8*)(p);
  u.h[1] = *(const bf16x8*)(p + 16);
  return u.v;
}
// B fragment (32x16 bf16), LDS stored [N][K]: lane -> N=lane&15,
// K = {half*8..half*8+7, 16+half*8..16+half*8+7}. p already offset by half*8.
__device__ __forceinline__ v16bf frag_b(const __bf16* p) {
  union { v16bf v; bf16x8 h[2]; } u;
  u.h[0] = *(const bf16x8*)(p);
  u.h[1] = *(const bf16x8*)(p + 16);
  return u.v;
}

__device__ __forceinline__ bf16x4 cvt4(float4 v) {
  bf16x4 pk = {(__bf16)v.x, (__bf16)v.y, (__bf16)v.z, (__bf16)v.w};
  return pk;
}

template <int AMODE, int EMODE>
__global__ __launch_bounds__(256) void kan_gemm(
    const float* __restrict__ Af,   // concat lhs / spline x source / feat1
    const float* __restrict__ Af2,  // concat rhs / feat2
    const __bf16* __restrict__ Abf, // bf16 A (silu activations)
    const float* __restrict__ W,    // weights, (N,K) row-major fp32
    const float* __restrict__ bias, // attn_b
    float* __restrict__ outF,       // f32 output (kan_in / tmp / h / d_out)
    __bf16* __restrict__ outBf,     // bf16 silu output
    const float* __restrict__ addF, // residual (base-path) input
    int K)                          // reduction depth (A-column units)
{
  __shared__ __bf16 sA[2][BM * LDT];
  __shared__ __bf16 sB[2][BN * LDT];

  const int tid  = threadIdx.x;
  const int lane = tid & 31;
  const int wave = tid >> 5;
  const int l16  = lane & 15;
  const int half = lane >> 4;
  const int wm   = wave & 3;   // 4 waves along M, 32 rows each
  const int wn   = wave >> 2;  // 2 waves along N, 64 cols each
  const int m0   = blockIdx.y * BM;
  const int n0   = blockIdx.x * BN;

  const int rowL = tid >> 1;        // 0..127: loader row
  const int koff = (tid & 1) * 16;  // loader half-row

  v8f zero = {0.f, 0.f, 0.f, 0.f, 0.f, 0.f, 0.f, 0.f};
  v8f acc[2][4];
#pragma unroll
  for (int i = 0; i < 2; ++i)
#pragma unroll
    for (int j = 0; j < 4; ++j) acc[i][j] = zero;

  // ---------------- prefetch registers ----------------
  bf16x4 ra[4];    // A tile staging (converted)
  bf16x8 rsp[2];   // spline bases staging
  bf16x8 rb16[2];  // bf16-A staging
  bf16x4 rw[4];    // B (weights) staging

  auto load_tile = [&](int kb) {
    // ---- A operand -> registers (convert early; stores later) ----
    if constexpr (AMODE == A_CONCAT) {
      const int col = kb + koff;  // 16-chunk never straddles the concat seam
      const float* p = (col < FDIM)
                           ? (Af  + (size_t)(m0 + rowL) * FDIM + col)
                           : (Af2 + (size_t)(m0 + rowL) * FDIM + (col - FDIM));
#pragma unroll
      for (int j = 0; j < 4; ++j) ra[j] = cvt4(((const float4*)p)[j]);
    } else if constexpr (AMODE == A_BF16) {
      const __bf16* p = Abf + (size_t)(m0 + rowL) * K + kb + koff;
      rb16[0] = *(const bf16x8*)(p);
      rb16[1] = *(const bf16x8*)(p + 8);
    } else {  // A_SPLINE: BK=32 covers 4 input features x 8 bases
      const int i0 = kb >> 3;
      const int KX = K >> 3;
#pragma unroll
      for (int t = 0; t < 2; ++t) {
        int vi = tid * 2 + t;  // 0..511 x-values per tile
        int r  = vi >> 2;
        int fi = vi & 3;
        float x = Af[(size_t)(m0 + r) * KX + i0 + fi];
        float bs[8];
        bspline8(x, bs);
#pragma unroll
        for (int q = 0; q < 8; ++q) rsp[t][q] = (__bf16)bs[q];
      }
    }
    // ---- B (weights) -> registers ----
    const float* p = W + (size_t)(n0 + rowL) * K + kb + koff;
#pragma unroll
    for (int j = 0; j < 4; ++j) rw[j] = cvt4(((const float4*)p)[j]);
  };

  auto store_tile = [&](int buf) {
    if constexpr (AMODE == A_SPLINE) {
#pragma unroll
      for (int t = 0; t < 2; ++t) {
        int vi = tid * 2 + t;
        int r  = vi >> 2;
        int fi = vi & 3;
        *(bf16x8*)&sA[buf][r * LDT + fi * 8] = rsp[t];
      }
    } else if constexpr (AMODE == A_BF16) {
      *(bf16x8*)&sA[buf][rowL * LDT + koff]     = rb16[0];
      *(bf16x8*)&sA[buf][rowL * LDT + koff + 8] = rb16[1];
    } else {
#pragma unroll
      for (int j = 0; j < 4; ++j)
        *(bf16x4*)&sA[buf][rowL * LDT + koff + j * 4] = ra[j];
    }
#pragma unroll
    for (int j = 0; j < 4; ++j)
      *(bf16x4*)&sB[buf][rowL * LDT + koff + j * 4] = rw[j];
  };

  const int nt = K / BK;
  load_tile(0);
  store_tile(0);
  __syncthreads();

  for (int t = 0; t < nt; ++t) {
    const int cur = t & 1;
    if (t + 1 < nt) load_tile((t + 1) * BK);  // overlap global with WMMA

    v16bf afrag[2], bfrag[4];
    const __bf16* abase = &sA[cur][(wm * 32 + l16) * LDT + half * 8];
    afrag[0] = frag_a(abase);
    afrag[1] = frag_a(abase + 16 * LDT);
    const __bf16* bbase = &sB[cur][(wn * 64 + l16) * LDT + half * 8];
#pragma unroll
    for (int j = 0; j < 4; ++j) bfrag[j] = frag_b(bbase + j * 16 * LDT);
#pragma unroll
    for (int i = 0; i < 2; ++i)
#pragma unroll
      for (int j = 0; j < 4; ++j)
        acc[i][j] = __builtin_amdgcn_wmma_f32_16x16x32_bf16(
            false, afrag[i], false, bfrag[j], (short)0, acc[i][j], false, false);

    if (t + 1 < nt) store_tile(cur ^ 1);
    __syncthreads();
  }

  // ---------------- epilogue (32-bit indexing) ----------------
#pragma unroll
  for (int i = 0; i < 2; ++i) {
#pragma unroll
    for (int j = 0; j < 4; ++j) {
#pragma unroll
      for (int v = 0; v < 8; ++v) {
        unsigned row = (unsigned)(m0 + wm * 32 + i * 16 + v + half * 8);
        unsigned col = (unsigned)(n0 + wn * 64 + j * 16 + l16);
        float val = acc[i][j][v];
        if constexpr (EMODE == E_ATTN) {
          float att = sigmoidf_(val + bias[col]);
          float f1 = Af [row * (unsigned)FDIM + col];
          float f2 = Af2[row * (unsigned)FDIM + col];
          float k1 = f1 * att;
          float k2 = f2 * (1.0f - att);
          unsigned o1 = row * (unsigned)(2 * FDIM) + col;
          unsigned o2 = o1 + FDIM;
          outF[o1] = k1;                  outF[o2] = k2;
          outBf[o1] = (__bf16)siluf_(k1); outBf[o2] = (__bf16)siluf_(k2);
        } else if constexpr (EMODE == E_PLAIN) {
          outF[row * (unsigned)FDIM + col] = val;
        } else if constexpr (EMODE == E_SPLINE0) {
          unsigned o = row * (unsigned)FDIM + col;
          float hv = val + addF[o];
          outF[o]  = hv;
          outBf[o] = (__bf16)siluf_(hv);
        } else {  // E_SPLINE1
          unsigned o = row * (unsigned)FDIM + col;
          outF[o] = val + addF[o];
        }
      }
    }
  }
}

extern "C" void kernel_launch(void* const* d_in, const int* in_sizes, int n_in,
                              void* d_out, int out_size, void* d_ws, size_t ws_size,
                              hipStream_t stream) {
  const float* feat1     = (const float*)d_in[0];
  const float* feat2     = (const float*)d_in[1];
  const float* attn_w    = (const float*)d_in[2];  // (F, 2F)    -> [N][K]
  const float* attn_b    = (const float*)d_in[3];
  const float* base_w0   = (const float*)d_in[4];  // (F, 2F)
  const float* spline_w0 = (const float*)d_in[5];  // (F, 2F, 8) -> [N][K=16384]
  const float* base_w1   = (const float*)d_in[6];  // (F, F)
  const float* spline_w1 = (const float*)d_in[7];  // (F, F, 8)  -> [N][K=8192]
  float* out = (float*)d_out;

  char* ws = (char*)d_ws;
  float*  kan_in = (float*) (ws);                          // 8192*2048 f32 = 64MB
  __bf16* silu0  = (__bf16*)(ws + ((size_t)64  << 20));    // 8192*2048 bf16 = 32MB
  float*  tmp    = (float*) (ws + ((size_t)96  << 20));    // 8192*1024 f32 = 32MB
  float*  hbuf   = (float*) (ws + ((size_t)128 << 20));    // 8192*1024 f32 = 32MB
  __bf16* siluh  = (__bf16*)(ws + ((size_t)160 << 20));    // 8192*1024 bf16 = 16MB

  dim3 blk(256);
  dim3 grd(FDIM / BN, BATCH / BM);  // (8, 64)

  // 1) attention = sigmoid(concat @ attn_w^T + b); gate -> kan_in f32 + silu bf16
  kan_gemm<A_CONCAT, E_ATTN><<<grd, blk, 0, stream>>>(
      feat1, feat2, nullptr, attn_w, attn_b, kan_in, silu0, nullptr, 2 * FDIM);
  // 2) base0: silu(kan_in) @ base_w0^T -> tmp
  kan_gemm<A_BF16, E_PLAIN><<<grd, blk, 0, stream>>>(
      nullptr, nullptr, silu0, base_w0, nullptr, tmp, nullptr, nullptr, 2 * FDIM);
  // 3) spline0: bases(kan_in) @ spline_w0^T + tmp -> h, silu(h)
  kan_gemm<A_SPLINE, E_SPLINE0><<<grd, blk, 0, stream>>>(
      kan_in, nullptr, nullptr, spline_w0, nullptr, hbuf, siluh, tmp, 16 * FDIM);
  // 4) base1: silu(h) @ base_w1^T -> tmp
  kan_gemm<A_BF16, E_PLAIN><<<grd, blk, 0, stream>>>(
      nullptr, nullptr, siluh, base_w1, nullptr, tmp, nullptr, nullptr, FDIM);
  // 5) spline1: bases(h) @ spline_w1^T + tmp -> d_out
  kan_gemm<A_SPLINE, E_SPLINE1><<<grd, blk, 0, stream>>>(
      hbuf, nullptr, nullptr, spline_w1, nullptr, out, nullptr, tmp, 8 * FDIM);

  (void)in_sizes; (void)n_in; (void)out_size; (void)ws_size;
}
